// DPMPModel_58609123721543
// MI455X (gfx1250) — matrix operand order repaired and checked by reading, not compiled, not run on hardware
//
#include <hip/hip_runtime.h>

// ---------------- static model configuration ----------------
#define NATOMS   96
#define T0       48           // atoms per type
#define KIMG     27
#define MNEI     (NATOMS*KIMG)    // 2592
#define NMTOT    (NATOMS*MNEI)    // 248832
#define MSPLIT   (T0*KIMG)        // 1296
#define RCUT_F   6.0f
#define SRMEAN_F 0.05f
#define SRSTD_F  0.12f
#define XRSTD_F  0.06f
#define NORM_F   40.0f
#define PI_F     3.14159265358979f
#define TILES_PER_N (MNEI/16)     // 162
#define NTILES   (NATOMS*TILES_PER_N) // 15552

typedef __attribute__((ext_vector_type(16))) _Float16 v16h;
typedef __attribute__((ext_vector_type(8)))  float    v8f;

// Native v_tanh_f32 on gfx1250 (transcendental VALU per CDNA5 ISA §8.4);
// guarded so compile never breaks if the builtin name is absent.
#if __has_builtin(__builtin_amdgcn_tanh_f32)
#define FAST_TANH(x) __builtin_amdgcn_tanh_f32(x)
#elif __has_builtin(__builtin_amdgcn_tanhf)
#define FAST_TANH(x) __builtin_amdgcn_tanhf(x)
#else
#define FAST_TANH(x) tanhf(x)
#endif

// ---------------- WMMA helpers ----------------
// A fragment (16x32 f16): lane = row + 16*half, element e=2v+o; k = (v<4?2v+8h:16+2(v-4)+8h)+o.
// Inverse map (k -> half,elem):  half=(k>>3)&1 ; elem=(k&7)|((k&16)>>1).
// B fragment (32x16 f16): lane = col + 16*half; k = kt*32 + 8*half + (elem&7) + 16*(elem>>3).
__device__ __forceinline__ v8f wmma16x16x32(v16h a, v16h b, v8f c) {
    return __builtin_amdgcn_wmma_f32_16x16x32_f16(false, a, false, b, (short)0, c, false, false);
}
__device__ __forceinline__ void fragposA(int k, int& half, int& e) {
    half = (k >> 3) & 1;
    e = (k & 7) | ((k & 16) >> 1);
}

// ---------------- parameter pointer bundles ----------------
struct SmallNets {           // nets: 0=embedI 1=embedJ 2=embed ; pair = i*2+j
    const float* b0[3][4]; const float* w0[3][4];
    const float* b1[3][4]; const float* w1[3][4];
    const float* e2b0[4];  const float* e2w0[4];
    const float* e2b1[4];  const float* e2w1[4];
};
struct MPParams {
    const _Float16* W1f;     // [4][3ct][7kt][32][16] swizzled B fragments
    const _Float16* W2f;     // [4][6ct][2kt][32][16]
    const float* b1[4];      // layer0 bias (48)
    const float* b2[4];      // layer1 bias (96)
};
struct FitParams {
    const _Float16 *W0, *W1, *W2;   // swizzled: [15][12][512], [15][8][512], [15][8][512]
    const float *b0, *b1, *dt1, *b2, *dt2, *bF, *wF, *Eb;
};
struct FitBoth { FitParams t[2]; };

// ---------------- K0: zero accumulators ----------------
__global__ void zero_kernel(float* __restrict__ G2, float* __restrict__ out) {
    int i = blockIdx.x*blockDim.x + threadIdx.x;
    if (i < 4*96*96) G2[i] = 0.f;
    if (i == 0) out[0] = 0.f;
}

// ---------------- K1: f32 weights -> f16 B-fragment-major layout ----------------
// dst layout: [Nc/16][KP/32][lane(32)][elem(16)] ; one wave B-fragment = 32 contiguous bytes/lane.
__global__ void cvt_swz_frag(const float* __restrict__ src, _Float16* __restrict__ dst,
                             int K, int Nc, int KP) {
    int i = blockIdx.x*blockDim.x + threadIdx.x;
    int nkt = KP >> 5, nct = Nc >> 4;
    if (i >= nct*nkt*512) return;
    int e    = i & 15;
    int lane = (i >> 4) & 31;
    int ft   = i >> 9;
    int ct = ft / nkt, kt = ft % nkt;
    int k = kt*32 + ((lane >> 4) << 3) + (e & 7) + ((e & 8) << 1);     // kt*32 + 8*half + (elem&7) + 16*(elem>>3)
    int c = ct*16 + (lane & 15);
    dst[i] = (k < K) ? (_Float16)src[k*Nc + c] : (_Float16)0.f;
}

// ---------------- K2: geometry: r, sr, R ----------------
__global__ void geom_kernel(const float* __restrict__ coord, const float* __restrict__ box,
                            float* __restrict__ Rbuf, float* __restrict__ srbias,
                            float* __restrict__ srnorm) {
    int idx = blockIdx.x*blockDim.x + threadIdx.x;
    if (idx >= NMTOT) return;
    int n = idx / MNEI, m = idx % MNEI;
    int jat = m / KIMG, s = m % KIMG;
    float si = (float)(s/9 - 1), sj = (float)((s/3)%3 - 1), sk = (float)(s%3 - 1);
    float x[3]; float r2 = 0.f;
    #pragma unroll
    for (int d = 0; d < 3; ++d) {
        float shift = si*box[0*3+d] + sj*box[1*3+d] + sk*box[2*3+d];
        x[d] = coord[d*NATOMS + jat] + shift - coord[d*NATOMS + n];
        r2 = fmaf(x[d], x[d], r2);
    }
    float r = sqrtf(r2);
    float rs = (r < 1e-6f) ? 1e6f : r;
    float sr = 0.f;
    if (rs < RCUT_F) sr = 0.5f*(cosf(PI_F*rs/RCUT_F) + 1.f)/rs;
    srbias[idx] = sr / SRSTD_F;
    srnorm[idx] = (sr - SRMEAN_F) / SRSTD_F;
    float rsr = (sr / rs) / XRSTD_F + 1e-15f;   // E3NORM = 1
    #pragma unroll
    for (int d = 0; d < 3; ++d) Rbuf[(size_t)d*NMTOT + idx] = rsr * x[d];
}

// ---------------- K3: small embedding nets (scalar input; VALU) ----------------
__global__ void __launch_bounds__(256)
smallnets_kernel(const float* __restrict__ srnorm,
                 _Float16* __restrict__ embI, _Float16* __restrict__ embJ,
                 _Float16* __restrict__ embE, _Float16* __restrict__ emb2,
                 SmallNets P) {
    int idx = blockIdx.x*blockDim.x + threadIdx.x;
    if (idx >= NMTOT) return;
    int n = idx / MNEI, m = idx % MNEI;
    int pair = (n >= T0)*2 + (m >= MSPLIT);
    float s = srnorm[idx];
    _Float16* outs[3] = {embI, embJ, embE};
    for (int net = 0; net < 3; ++net) {
        const float* b0 = P.b0[net][pair]; const float* w0 = P.w0[net][pair];
        const float* b1 = P.b1[net][pair]; const float* w1 = P.w1[net][pair];
        float h0[16];
        #pragma unroll
        for (int k = 0; k < 16; ++k) h0[k] = FAST_TANH(fmaf(s, w0[k], b0[k]));
        v16h t0 = {}, t1 = {};
        #pragma unroll
        for (int c = 0; c < 32; ++c) {
            float acc = b1[c];
            #pragma unroll
            for (int k = 0; k < 16; ++k) acc = fmaf(h0[k], w1[k*32 + c], acc);
            float val = FAST_TANH(acc) + h0[c >> 1];       // residual repeat(x,2)
            if (c < 16) t0[c] = (_Float16)val; else t1[c - 16] = (_Float16)val;
        }
        v16h* o = (v16h*)(outs[net] + (size_t)idx*32);
        o[0] = t0; o[1] = t1;
    }
    { // embed2: 1 -> 8 -> 16
        const float* b0 = P.e2b0[pair]; const float* w0 = P.e2w0[pair];
        const float* b1 = P.e2b1[pair]; const float* w1 = P.e2w1[pair];
        float h0[8];
        #pragma unroll
        for (int k = 0; k < 8; ++k) h0[k] = FAST_TANH(fmaf(s, w0[k], b0[k]));
        v16h t0 = {};
        #pragma unroll
        for (int c = 0; c < 16; ++c) {
            float acc = b1[c];
            #pragma unroll
            for (int k = 0; k < 8; ++k) acc = fmaf(h0[k], w1[k*16 + c], acc);
            t0[c] = (_Float16)(FAST_TANH(acc) + h0[c >> 1]);
        }
        *(v16h*)(emb2 + (size_t)idx*16) = t0;
    }
}

// ---------------- K4: per-atom G reductions over M ----------------
// G4 logical layout: [net(3)][a(4: inv,x,y,z)][96][32]
__global__ void __launch_bounds__(256)
greduce_kernel(const _Float16* __restrict__ embI, const _Float16* __restrict__ embJ,
               const _Float16* __restrict__ embE, const float* __restrict__ srbias,
               const float* __restrict__ Rbuf, const float* __restrict__ GGbias,
               float* __restrict__ G4) {
    __shared__ float red[4][8][32];
    int n = blockIdx.x / 3, net = blockIdx.x % 3;
    const _Float16* E = (net == 0) ? embI : ((net == 1) ? embJ : embE);
    int c = threadIdx.x & 31, g = threadIdx.x >> 5;
    float a0 = 0.f, a1 = 0.f, a2 = 0.f, a3 = 0.f;
    for (int m = g; m < MNEI; m += 8) {
        size_t idx = (size_t)n*MNEI + m;
        float e = (float)E[idx*32 + c];
        a0 = fmaf(srbias[idx], e, a0);
        a1 = fmaf(Rbuf[idx], e, a1);
        a2 = fmaf(Rbuf[(size_t)NMTOT + idx], e, a2);
        a3 = fmaf(Rbuf[2*(size_t)NMTOT + idx], e, a3);
    }
    red[0][g][c] = a0; red[1][g][c] = a1; red[2][g][c] = a2; red[3][g][c] = a3;
    __syncthreads();
    if (g == 0) {
        #pragma unroll
        for (int a = 0; a < 4; ++a) {
            float t = 0.f;
            #pragma unroll
            for (int gg = 0; gg < 8; ++gg) t += red[a][gg][c];
            t /= NORM_F;
            if (net == 2 && a == 0) t += GGbias[c];
            G4[((net*4 + a)*96 + n)*32 + c] = t;
        }
    }
}

// ---------------- K5: per-atom FII/FJJ outer products (64 each) ----------------
__global__ void ff_kernel(const float* __restrict__ G4, float* __restrict__ FF) {
    int n = blockIdx.x, e = threadIdx.x;   // 64 threads: e = p*32 + c
    int p = e >> 5, c = e & 31;
    float t = 0.f;
    #pragma unroll
    for (int a = 0; a < 4; ++a) {
        const float* ga = G4 + (size_t)((8 + a)*96 + n)*32;   // net2 base
        t = fmaf(ga[c], ga[p], t);
    }
    FF[n*64 + e] = t;
}

// ---------------- K6: fused F-build + embedMP WMMA GEMM + output reduce ----------------
#define SHS 72    // 64 + pad
__global__ void __launch_bounds__(128)
mp_gemm_kernel(const _Float16* __restrict__ emb2, const float* __restrict__ G4,
               const float* __restrict__ FF, const float* __restrict__ Rbuf,
               const float* __restrict__ srbias, float* __restrict__ G2, MPParams P) {
    __shared__ __align__(16) _Float16 sFfrag[4][7][32][16];  // A fragments, layer 1
    __shared__ __align__(16) _Float16 sH[4][16*SHS];          // H row-major (residual + layer2 A)
    __shared__ float sRW[4][16][4];
    int w = threadIdx.x >> 5, lane = threadIdx.x & 31;
    int tile = blockIdx.x*4 + w;                 // 15552 tiles exactly
    int n  = tile / TILES_PER_N;
    int m0 = (tile % TILES_PER_N) * 16;
    int pair = (n >= T0)*2 + (m0 >= MSPLIT);
    const _Float16* W1 = P.W1f + (size_t)pair*(3*7*512);
    const _Float16* W2 = P.W2f + (size_t)pair*(6*2*512);
    const float* bias1 = P.b1[pair];
    const float* bias2 = P.b2[pair];
    __builtin_prefetch(W1, 0, 0);
    __builtin_prefetch(W2, 0, 0);
    if (lane < 16) {                              // per-row reduction weights
        size_t idx = (size_t)n*MNEI + m0 + lane;
        sRW[w][lane][0] = srbias[idx];
        sRW[w][lane][1] = Rbuf[idx];
        sRW[w][lane][2] = Rbuf[(size_t)NMTOT + idx];
        sRW[w][lane][3] = Rbuf[2*(size_t)NMTOT + idx];
    }
    // Build F tile (16 x 224) directly in A-fragment layout:
    // [embed2(16) | FI(32) | FJ(32) | FII(64) | FJJ(64) | pad16]
    for (int t = lane; t < 16*224; t += 32) {
        int rr = t / 224, cc = t % 224;
        int m = m0 + rr; size_t idx = (size_t)n*MNEI + m; int jat = m / KIMG;
        float val = 0.f;
        if (cc < 16) {
            val = (float)emb2[idx*16 + cc];
        } else if (cc < 48) {
            int c = cc - 16;
            const float* g = G4 + (size_t)((0*4 + 1)*96 + n)*32 + c;     // GIeqv
            val = sRW[w][rr][1]*g[0] + sRW[w][rr][2]*g[96*32] + sRW[w][rr][3]*g[2*96*32];
        } else if (cc < 80) {
            int c = cc - 48;
            const float* g = G4 + (size_t)((1*4 + 1)*96 + jat)*32 + c;   // GJeqv
            val = sRW[w][rr][1]*g[0] + sRW[w][rr][2]*g[96*32] + sRW[w][rr][3]*g[2*96*32];
        } else if (cc < 144) {
            val = FF[n*64 + (cc - 80)];
        } else if (cc < 208) {
            val = FF[jat*64 + (cc - 144)];
        }
        int kt = cc >> 5, k = cc & 31, h, e;
        fragposA(k, h, e);
        sFfrag[w][kt][rr + 16*h][e] = (_Float16)val;
    }
    for (int t = lane; t < 16*16; t += 32)        // zero K pad of H (48..63)
        sH[w][(t >> 4)*SHS + 48 + (t & 15)] = (_Float16)0.f;
    __syncthreads();
    int half = lane >> 4, ln = lane & 15;
    // Hoist all layer-1 A fragments (reused by 3 col tiles): 2x ds_load_b128 each.
    v16h afr[7];
    #pragma unroll
    for (int kt = 0; kt < 7; ++kt) afr[kt] = *(const v16h*)&sFfrag[w][kt][lane][0];
    // Layer 1: (16x224)@(224x48), tanh
    #pragma unroll
    for (int ct = 0; ct < 3; ++ct) {
        v8f acc = {};
        #pragma unroll
        for (int kt = 0; kt < 7; ++kt) {
            v16h b = *(const v16h*)(W1 + (size_t)((ct*7 + kt)*32 + lane)*16);
            acc = wmma16x16x32(afr[kt], b, acc);
        }
        #pragma unroll
        for (int v = 0; v < 8; ++v) {
            int row = v + 8*half, col = ct*16 + ln;
            sH[w][row*SHS + col] = (_Float16)FAST_TANH(acc[v] + bias1[col]);
        }
    }
    __syncthreads();
    // Layer-2 A fragments from H (scattered; only 2 k-tiles)
    v16h a2[2];
    #pragma unroll
    for (int kt = 0; kt < 2; ++kt) {
        v16h a = {};
        #pragma unroll
        for (int e = 0; e < 16; ++e) {
            int kin = (e & 7) + ((e & 8) << 1);           // inverse of elem map, half=0 part
            int k = kt*32 + kin + 8*half;
            a[e] = sH[w][ln*SHS + k];
        }
        a2[kt] = a;
    }
    // Layer 2: (16x64)@(64x96), residual repeat + fused reduction to G2
    #pragma unroll
    for (int ct = 0; ct < 6; ++ct) {
        v8f acc = {};
        #pragma unroll
        for (int kt = 0; kt < 2; ++kt) {
            v16h b = *(const v16h*)(W2 + (size_t)((ct*2 + kt)*32 + lane)*16);
            acc = wmma16x16x32(a2[kt], b, acc);
        }
        int col = ct*16 + ln;
        float p0 = 0.f, p1 = 0.f, p2 = 0.f, p3 = 0.f;
        #pragma unroll
        for (int v = 0; v < 8; ++v) {
            int row = v + 8*half;
            float e = FAST_TANH(acc[v] + bias2[col]) + (float)sH[w][row*SHS + (col >> 1)];
            p0 = fmaf(sRW[w][row][0], e, p0);
            p1 = fmaf(sRW[w][row][1], e, p1);
            p2 = fmaf(sRW[w][row][2], e, p2);
            p3 = fmaf(sRW[w][row][3], e, p3);
        }
        p0 += __shfl_xor(p0, 16, 32);
        p1 += __shfl_xor(p1, 16, 32);
        p2 += __shfl_xor(p2, 16, 32);
        p3 += __shfl_xor(p3, 16, 32);
        if (half == 0) {
            atomicAdd(&G2[(0*96 + n)*96 + col], p0);
            atomicAdd(&G2[(1*96 + n)*96 + col], p1);
            atomicAdd(&G2[(2*96 + n)*96 + col], p2);
            atomicAdd(&G2[(3*96 + n)*96 + col], p3);
        }
    }
}

// ---------------- K7: per-atom feature outer product -> A-fragment layout ----------------
// FeatFrag: [rowtile(6)][kt(12)][lane(32)][elem(16)] f16
__global__ void feat_kernel(const float* __restrict__ G2, const float* __restrict__ Gbias,
                            _Float16* __restrict__ FeatFrag) {
    __shared__ float g[4][96];
    int n = blockIdx.x;
    for (int t = threadIdx.x; t < 384; t += blockDim.x) {
        int a = t / 96, b = t % 96;
        float v = G2[(a*96 + n)*96 + b] / NORM_F;
        if (a == 0) v += Gbias[b];
        g[a][b] = v;
    }
    __syncthreads();
    for (int t = threadIdx.x; t < 384; t += blockDim.x) {
        int p = t / 96, b = t % 96;
        float f = 0.f;
        #pragma unroll
        for (int a = 0; a < 4; ++a) f = fmaf(g[a][b], g[a][p], f);
        int col = p*96 + b;                       // Feat column index
        int kt = col >> 5, k = col & 31, h, e;
        fragposA(k, h, e);
        size_t o = ((size_t)((n >> 4)*12 + kt)*32 + (n & 15) + 16*h)*16 + e;
        FeatFrag[o] = (_Float16)f;
    }
}

// ---------------- K8: fit net (WMMA) + energy sum ----------------
__device__ __forceinline__ void fit_layer(const _Float16* __restrict__ Xin,
                                          _Float16* __restrict__ Xout,
                                          const _Float16* __restrict__ W,   // swizzled [15][8][512]
                                          const float* b, const float* dt,
                                          int w, int lane) {
    int half = lane >> 4, ln = lane & 15;
    for (int job = w; job < 45; job += 4) {
        int rt = job / 15, ct = job % 15;
        v8f acc = {};
        #pragma unroll
        for (int kt = 0; kt < 8; ++kt) {
            v16h a = {};
            #pragma unroll
            for (int e = 0; e < 16; ++e) {
                int kin = (e & 7) + ((e & 8) << 1);
                a[e] = Xin[(rt*16 + ln)*256 + kt*32 + kin + 8*half];
            }
            v16h bb = *(const v16h*)(W + (size_t)((ct*8 + kt)*32 + lane)*16);
            acc = wmma16x16x32(a, bb, acc);
        }
        #pragma unroll
        for (int v = 0; v < 8; ++v) {
            int row = rt*16 + v + 8*half, col = ct*16 + ln;
            float x = (float)Xin[row*256 + col];
            Xout[row*256 + col] = (_Float16)(FAST_TANH(acc[v] + b[col])*dt[col] + x);
        }
    }
}

__global__ void __launch_bounds__(128)
fit_kernel(const _Float16* __restrict__ FeatFrag, float* __restrict__ out, FitBoth FP) {
    __shared__ __align__(16) _Float16 sX[2][48*256];
    int t = blockIdx.x;
    FitParams P = FP.t[t];
    int w = threadIdx.x >> 5, lane = threadIdx.x & 31, half = lane >> 4, ln = lane & 15;
    for (int i = threadIdx.x; i < 2*48*16; i += 128) {     // zero K pads (240..255)
        int buf = i / (48*16), r = (i >> 4) % 48, c = 240 + (i & 15);
        sX[buf][r*256 + c] = (_Float16)0.f;
    }
    __syncthreads();
    // Layer 0: A fragments straight from FeatFrag (2x b128/lane), B swizzled
    for (int job = w; job < 45; job += 4) {
        int rt = job / 15, ct = job % 15;
        v8f acc = {};
        #pragma unroll
        for (int kt = 0; kt < 12; ++kt) {
            v16h a = *(const v16h*)(FeatFrag + (size_t)(((t*3 + rt)*12 + kt)*32 + lane)*16);
            v16h bb = *(const v16h*)(P.W0 + (size_t)((ct*12 + kt)*32 + lane)*16);
            acc = wmma16x16x32(a, bb, acc);
        }
        #pragma unroll
        for (int v = 0; v < 8; ++v) {
            int row = rt*16 + v + 8*half, col = ct*16 + ln;
            sX[0][row*256 + col] = (_Float16)FAST_TANH(acc[v] + P.b0[col]);
        }
    }
    __syncthreads();
    fit_layer(sX[0], sX[1], P.W1, P.b1, P.dt1, w, lane);
    __syncthreads();
    fit_layer(sX[1], sX[0], P.W2, P.b2, P.dt2, w, lane);
    __syncthreads();
    if (threadIdx.x < 48) {                                // final 240->1 + Ebias
        int r = threadIdx.x;
        float acc = P.bF[0] + P.Eb[0];
        for (int c = 0; c < 240; ++c) acc = fmaf((float)sX[0][r*256 + c], P.wF[c], acc);
        atomicAdd(out, acc);
    }
}

// ---------------- host side ----------------
extern "C" void kernel_launch(void* const* d_in, const int* in_sizes, int n_in,
                              void* d_out, int out_size, void* d_ws, size_t ws_size,
                              hipStream_t stream) {
    (void)n_in; (void)out_size; (void)ws_size;
    auto f = [&](int i) { return (const float*)d_in[i]; };
    // Assumed flattening: coord_3N, box_33, then params pytree
    // (alphabetical dict keys: Ebias, GGbias, Gbias, embed, embed2, embedI, embedJ, embedMP, fit;
    //  inner layer dicts alphabetical: b, dt, w).
    const float* coord = f(0);
    const float* box   = f(1);
    if (in_sizes[0] == 9) { coord = f(1); box = f(0); }    // tolerate sorted top-level
    const int EBIAS = 2, GGB = 4, GB = 5;
    const int EMB = 6, EMB2 = 22, EMBI = 38, EMBJ = 54, EMBMP = 70, FIT = 86;

    // workspace carve-out (~62 MB)
    char* ws = (char*)d_ws; size_t off = 0;
    auto alloc = [&](size_t bytes) { void* p = ws + off; off = (off + bytes + 255) & ~(size_t)255; return p; };
    float*     Rbuf   = (float*)alloc((size_t)3*NMTOT*4);
    float*     srbias = (float*)alloc((size_t)NMTOT*4);
    float*     srnorm = (float*)alloc((size_t)NMTOT*4);
    _Float16*  embI   = (_Float16*)alloc((size_t)NMTOT*32*2);
    _Float16*  embJ   = (_Float16*)alloc((size_t)NMTOT*32*2);
    _Float16*  embE   = (_Float16*)alloc((size_t)NMTOT*32*2);
    _Float16*  emb2   = (_Float16*)alloc((size_t)NMTOT*16*2);
    float*     G4     = (float*)alloc((size_t)3*4*96*32*4);
    float*     FF     = (float*)alloc((size_t)96*64*4);
    float*     G2     = (float*)alloc((size_t)4*96*96*4);
    _Float16*  FeatFr = (_Float16*)alloc((size_t)6*12*32*16*2);
    _Float16*  W1f    = (_Float16*)alloc((size_t)4*3*7*512*2);
    _Float16*  W2f    = (_Float16*)alloc((size_t)4*6*2*512*2);
    _Float16*  fW0    = (_Float16*)alloc((size_t)2*15*12*512*2);
    _Float16*  fW1    = (_Float16*)alloc((size_t)2*15*8*512*2);
    _Float16*  fW2    = (_Float16*)alloc((size_t)2*15*8*512*2);

    zero_kernel<<<144, 256, 0, stream>>>(G2, (float*)d_out);

    for (int p = 0; p < 4; ++p) {
        cvt_swz_frag<<<(3*7*512+255)/256, 256, 0, stream>>>(f(EMBMP+4*p+1), W1f + (size_t)p*3*7*512, 208, 48, 224);
        cvt_swz_frag<<<(6*2*512+255)/256, 256, 0, stream>>>(f(EMBMP+4*p+3), W2f + (size_t)p*6*2*512,  48, 96,  64);
    }
    for (int t = 0; t < 2; ++t) {
        cvt_swz_frag<<<(15*12*512+255)/256, 256, 0, stream>>>(f(FIT+10*t+1), fW0 + (size_t)t*15*12*512, 384, 240, 384);
        cvt_swz_frag<<<(15*8*512+255)/256, 256, 0, stream>>>(f(FIT+10*t+4), fW1 + (size_t)t*15*8*512, 240, 240, 256);
        cvt_swz_frag<<<(15*8*512+255)/256, 256, 0, stream>>>(f(FIT+10*t+7), fW2 + (size_t)t*15*8*512, 240, 240, 256);
    }

    geom_kernel<<<NMTOT/256, 256, 0, stream>>>(coord, box, Rbuf, srbias, srnorm);

    SmallNets SN;
    const int base[3] = {EMBI, EMBJ, EMB};   // nets: 0=embedI 1=embedJ 2=embed
    for (int net = 0; net < 3; ++net)
        for (int p = 0; p < 4; ++p) {
            SN.b0[net][p] = f(base[net]+4*p+0); SN.w0[net][p] = f(base[net]+4*p+1);
            SN.b1[net][p] = f(base[net]+4*p+2); SN.w1[net][p] = f(base[net]+4*p+3);
        }
    for (int p = 0; p < 4; ++p) {
        SN.e2b0[p] = f(EMB2+4*p+0); SN.e2w0[p] = f(EMB2+4*p+1);
        SN.e2b1[p] = f(EMB2+4*p+2); SN.e2w1[p] = f(EMB2+4*p+3);
    }
    smallnets_kernel<<<NMTOT/256, 256, 0, stream>>>(srnorm, embI, embJ, embE, emb2, SN);

    greduce_kernel<<<NATOMS*3, 256, 0, stream>>>(embI, embJ, embE, srbias, Rbuf, f(GGB), G4);
    ff_kernel<<<NATOMS, 64, 0, stream>>>(G4, FF);

    MPParams MP; MP.W1f = W1f; MP.W2f = W2f;
    for (int p = 0; p < 4; ++p) { MP.b1[p] = f(EMBMP+4*p+0); MP.b2[p] = f(EMBMP+4*p+2); }
    mp_gemm_kernel<<<NTILES/4, 128, 0, stream>>>(emb2, G4, FF, Rbuf, srbias, G2, MP);

    feat_kernel<<<NATOMS, 128, 0, stream>>>(G2, f(GB), FeatFr);

    FitBoth FB;
    for (int t = 0; t < 2; ++t) {
        int B = FIT + 10*t;
        FB.t[t].W0 = fW0 + (size_t)t*15*12*512;
        FB.t[t].W1 = fW1 + (size_t)t*15*8*512;
        FB.t[t].W2 = fW2 + (size_t)t*15*8*512;
        FB.t[t].b0 = f(B+0); FB.t[t].b1 = f(B+2); FB.t[t].dt1 = f(B+3);
        FB.t[t].b2 = f(B+5); FB.t[t].dt2 = f(B+6);
        FB.t[t].bF = f(B+8); FB.t[t].wF = f(B+9);
        FB.t[t].Eb = f(EBIAS + t);
    }
    fit_kernel<<<2, 128, 0, stream>>>(FeatFr, (float*)d_out, FB);
}
